// PAM_Module_7318624272662
// MI455X (gfx1250) — compile-verified, matrix operand order checked by reading, not run on hardware
//
#include <hip/hip_runtime.h>
#include <cmath>

// ---------------------------------------------------------------------------
// PAM module for MI455X (gfx1250). bf16 WMMA (f32 accumulate) for all 1x1-conv
// GEMMs, fed by 128-bit fragment loads:
//   A-fragments : 2 x global b128 from row-major bf16 weights
//   B-fragments : 2 x ds b128 from pixel-major [p][c] LDS tiles (stride 144B,
//                 36 dwords -> conflict-free across the 16 fragment lanes)
//   epilogues   : 8 consecutive channels per lane -> 1 x ds_store_b128
// Depthwise convs & KV contraction work on channel pairs (b32 LDS traffic).
// Deterministic: per-tile partial KV/Ksum reduced in fixed order (no atomics).
// ---------------------------------------------------------------------------

typedef unsigned short u16;
typedef __attribute__((ext_vector_type(16))) __bf16 v16bf;
typedef __attribute__((ext_vector_type(8)))  float  v8f;

union BF16x16 { v16bf v; uint4 q[2]; u16 s[16]; };

// ---- geometry -------------------------------------------------------------
#define BATCH 4
#define C64   64
#define M8    8
#define HH    224
#define WW    224
#define NPIX  (HH * WW)        // 50176
#define TH    8                // tile height (center)
#define TW    32               // tile width  (center)
#define HALO  2
#define PH    (TH + 2 * HALO)  // 12
#define PW    (TW + 2 * HALO)  // 36
#define PPIX  (PH * PW)        // 432 = 27 * 16
#define CEN   (TH * TW)        // 256
#define XSTR  72               // per-pixel stride in u16 (144 B = 9 uint4)
#define XSTRW (XSTR / 2)       // 36 dwords per pixel
#define TILES_X (WW / TW)      // 7
#define TILES_Y (HH / TH)      // 28
#define NTILES  (TILES_X * TILES_Y)  // 196
#define NTHR  512              // 16 waves

// ---- bf16 helpers ---------------------------------------------------------
// exact RNE (used once per launch on weights)
__device__ __forceinline__ u16 f2bf_rne(float f) {
  union { float f; unsigned u; } a; a.f = f;
  unsigned r = a.u + 0x7FFFu + ((a.u >> 16) & 1u);
  return (u16)(r >> 16);
}
// fast round-half-up pack of two activations into one dword (bounded inputs)
__device__ __forceinline__ unsigned pk_bf(float lo, float hi) {
  union { float f; unsigned u; } a, b; a.f = lo; b.f = hi;
  unsigned al = a.u + 0x8000u;
  unsigned bh = b.u + 0x8000u;
  return (bh & 0xffff0000u) | (al >> 16);
}
// unpack channel pair from one dword
__device__ __forceinline__ float bf_lo(unsigned d) {
  union { unsigned u; float f; } a; a.u = d << 16; return a.f;
}
__device__ __forceinline__ float bf_hi(unsigned d) {
  union { unsigned u; float f; } a; a.u = d & 0xffff0000u; return a.f;
}
__device__ __forceinline__ float relu6f(float v) {
  return fminf(fmaxf(v, 0.f), 6.f);
}

// ---- WMMA (CDNA5: V_WMMA_F32_16X16X32_BF16, wave32) -----------------------
__device__ __forceinline__ v8f wmma_bf16(v16bf a, v16bf b, v8f c) {
  return __builtin_amdgcn_wmma_f32_16x16x32_bf16(false, a, false, b,
                                                 (short)0, c, false, false);
}

// A fragment (16 rows x 32 K) from row-major bf16 [rows][64] in global memory.
__device__ __forceinline__ v16bf load_a_g(const u16* __restrict__ w,
                                          int m0, int k0) {
  int lane = threadIdx.x & 31;
  int row  = m0 + (lane & 15);
  int hi   = lane >> 4;
  const uint4* p = (const uint4*)w;
  int base = row * 8 + (k0 >> 3) + hi;
  BF16x16 f;
  f.q[0] = p[base];
  f.q[1] = p[base + 2];
  return f.v;
}

// B fragment (32 K x 16 cols) from pixel-major LDS tile [pix][72 u16].
__device__ __forceinline__ v16bf load_b_l(const u16* xs, int n0, int k0) {
  int lane = threadIdx.x & 31;
  int n    = n0 + (lane & 15);
  int hi   = lane >> 4;
  const uint4* p = (const uint4*)xs;
  int base = n * 9 + (k0 >> 3) + hi;
  BF16x16 f;
  f.q[0] = p[base];
  f.q[1] = p[base + 2];
  return f.v;
}

// ===========================================================================
// Kernel 0: weight conversion f32->bf16 (RNE; q/k zero-padded to 16 rows) and
// BN folding. bnaff: [bn 0..3][scale[64] | bias[64]] (fc1, conv1, conv2, fc2)
// ===========================================================================
__global__ void pam_prep(const float* __restrict__ qw, const float* __restrict__ kw,
                         const float* __restrict__ fc1w, const float* __restrict__ fc2w,
                         const float* __restrict__ vw,
                         const float* __restrict__ bn0, const float* __restrict__ bn1,
                         const float* __restrict__ bn2, const float* __restrict__ bn3,
                         u16* wq, u16* wk, u16* w1, u16* w2, u16* wv, float* bnaff) {
  int tid = threadIdx.x;
  for (int i = tid; i < 4096; i += 256) {
    w1[i] = f2bf_rne(fc1w[i]);
    w2[i] = f2bf_rne(fc2w[i]);
    wv[i] = f2bf_rne(vw[i]);
  }
  for (int i = tid; i < 1024; i += 256) {   // 16 rows, rows 8..15 zero
    wq[i] = (i < 512) ? f2bf_rne(qw[i]) : (u16)0;
    wk[i] = (i < 512) ? f2bf_rne(kw[i]) : (u16)0;
  }
  {
    int bn = tid >> 6, c = tid & 63;
    const float* p = (bn == 0) ? bn0 : (bn == 1) ? bn1 : (bn == 2) ? bn2 : bn3;
    float s = p[c], bi = p[64 + c], m = p[128 + c], v = p[192 + c];
    float inv = s * rsqrtf(v + 1e-5f);
    bnaff[bn * 128 + c]      = inv;
    bnaff[bn * 128 + 64 + c] = bi - m * inv;
  }
}

// ===========================================================================
// Kernel 1: fused tile stage.  grid (7, 28, 4), block 512 (16 waves)
// ===========================================================================
__global__ __launch_bounds__(NTHR, 1) void pam_stage_a(
    const float* __restrict__ x,
    const u16* __restrict__ wq, const u16* __restrict__ wk,
    const u16* __restrict__ w1, const u16* __restrict__ w2,
    const u16* __restrict__ wv,
    const float* __restrict__ bnaff,
    const float* __restrict__ qb, const float* __restrict__ kb,
    const float* __restrict__ vb,
    const float* __restrict__ dw5, const float* __restrict__ dw3,
    float* __restrict__ Qout, float* __restrict__ pKV, float* __restrict__ pKs) {
  const int tid  = threadIdx.x;
  const int lane = tid & 31;
  const int wave = tid >> 5;
  const int bx = blockIdx.x, by = blockIdx.y, bz = blockIdx.z;
  const int tileIdx = by * TILES_X + bx;

  __shared__ __align__(16) u16 lx[PPIX * XSTR];      // x tile (halo)  62.2 KB
  __shared__ __align__(16) u16 lh[PPIX * XSTR];      // h / V,K alias  62.2 KB
  __shared__ __align__(16) u16 ls[CEN * XSTR];       // x1+x2 center   36.9 KB
  __shared__ float ldw[C64 * 25 + C64 * 9];          // dw weights      8.7 KB
  __shared__ float lbn[512 + C64 + 2 * M8];          // BN affine + biases

  if (tid == 0) {  // CDNA5 global_prefetch of hot weight matrices
    __builtin_prefetch(w1, 0, 0);
    __builtin_prefetch(w2, 0, 0);
    __builtin_prefetch(wv, 0, 0);
  }

  // ---- A) x tile with halo -> bf16, pixel-major [p][c] (b32 stores) ----
  for (int i = tid; i < 32 * PPIX; i += NTHR) {
    int c2 = i / PPIX, p = i - c2 * PPIX;        // c2: channel pair
    int rp = p / PW, cp = p - rp * PW;
    int gh = by * TH + rp - HALO;
    int gw = bx * TW + cp - HALO;
    float v0 = 0.f, v1 = 0.f;
    if ((unsigned)gh < (unsigned)HH && (unsigned)gw < (unsigned)WW) {
      size_t g = (((size_t)bz * C64 + 2 * c2) * HH + gh) * WW + gw;
      v0 = x[g];
      v1 = x[g + (size_t)HH * WW];
    }
    ((unsigned*)lx)[p * XSTRW + c2] = pk_bf(v0, v1);
  }
  for (int i = tid; i < C64 * 25 + C64 * 9; i += NTHR)
    ldw[i] = (i < C64 * 25) ? dw5[i] : dw3[i - C64 * 25];
  for (int i = tid; i < 512; i += NTHR) lbn[i] = bnaff[i];
  if (tid < C64) lbn[512 + tid] = vb[tid];
  if (tid < M8)  lbn[512 + C64 + tid] = qb[tid];
  if (tid < M8)  lbn[512 + C64 + M8 + tid] = kb[tid];
  __syncthreads();

  const float* sc0 = lbn + 0,   *bi0 = lbn + 64;
  const float* sc1 = lbn + 128, *bi1 = lbn + 192;
  const float* sc2 = lbn + 256, *bi2 = lbn + 320;
  const float* sc3 = lbn + 384, *bi3 = lbn + 448;
  const float* lvb = lbn + 512;
  const float* lqb = lbn + 512 + C64;
  const float* lkb = lbn + 512 + C64 + M8;

  // ---- B) fc1 GEMM: h = relu6(bn0(W1 @ x)) over 432 padded pixels ----
  for (int t = wave; t < 4 * 27; t += 16) {
    int mt = t / 27, nt = t - mt * 27;
    v8f acc = {};
    acc = wmma_bf16(load_a_g(w1, mt * 16, 0),  load_b_l(lx, nt * 16, 0),  acc);
    acc = wmma_bf16(load_a_g(w1, mt * 16, 32), load_b_l(lx, nt * 16, 32), acc);
    int hi = lane >> 4, col = lane & 15;
    int chb = mt * 16 + hi * 8;                 // 8 consecutive channels
    int p   = nt * 16 + col;
    float r[8];
#pragma unroll
    for (int i = 0; i < 8; ++i)
      r[i] = relu6f(acc[i] * sc0[chb + i] + bi0[chb + i]);
    uint4 o = { pk_bf(r[0], r[1]), pk_bf(r[2], r[3]),
                pk_bf(r[4], r[5]), pk_bf(r[6], r[7]) };
    *(uint4*)(lh + p * XSTR + chb) = o;         // ds_store_b128
  }
  __syncthreads();

  // ---- C) depthwise 5x5 & 3x3 on channel pairs (b32 LDS loads) ----
  for (int i = tid; i < 32 * CEN; i += NTHR) {
    int c2 = i & 31, c = c2 * 2, pc = i >> 5, r = pc >> 5, wc = pc & 31;
    const unsigned* hb = (const unsigned*)lh + c2;
    const float* k5a = ldw + c * 25;
    const float* k5b = k5a + 25;
    float s5a = 0.f, s5b = 0.f;
#pragma unroll
    for (int dy = 0; dy < 5; ++dy)
#pragma unroll
      for (int dx = 0; dx < 5; ++dx) {
        unsigned d = hb[((r + dy) * PW + wc + dx) * XSTRW];
        float wa = k5a[dy * 5 + dx], wb = k5b[dy * 5 + dx];
        s5a += wa * bf_lo(d);
        s5b += wb * bf_hi(d);
      }
    const float* k3a = ldw + C64 * 25 + c * 9;
    const float* k3b = k3a + 9;
    float s3a = 0.f, s3b = 0.f;
#pragma unroll
    for (int dy = 0; dy < 3; ++dy)
#pragma unroll
      for (int dx = 0; dx < 3; ++dx) {
        unsigned d = hb[((r + 1 + dy) * PW + wc + 1 + dx) * XSTRW];
        float wa = k3a[dy * 3 + dx], wb = k3b[dy * 3 + dx];
        s3a += wa * bf_lo(d);
        s3b += wb * bf_hi(d);
      }
    float ra = relu6f(s5a * sc1[c] + bi1[c])     + relu6f(s3a * sc2[c] + bi2[c]);
    float rb = relu6f(s5b * sc1[c + 1] + bi1[c + 1]) +
               relu6f(s3b * sc2[c + 1] + bi2[c + 1]);
    ((unsigned*)ls)[pc * XSTRW + c2] = pk_bf(ra, rb);
  }
  __syncthreads();  // lh dead from here; alias V and K into it

  u16*   lv = lh;                             // V [256 pix][72 u16] (36.9 KB)
  float* lk = (float*)(lh + CEN * XSTR);      // K [8][260] f32       (8.3 KB)

  // ---- D1) V = (Wv @ x_center) + v_b + relu6(bn3(W2 @ s)) ----
  for (int t = wave; t < 4 * 16; t += 16) {
    int mt = t >> 4, nt = t & 15;
    int pb = ((nt >> 1) + HALO) * PW + HALO + (nt & 1) * 16;  // center->padded
    v8f a2 = {}, av = {};
    a2 = wmma_bf16(load_a_g(w2, mt * 16, 0),  load_b_l(ls, nt * 16, 0),  a2);
    a2 = wmma_bf16(load_a_g(w2, mt * 16, 32), load_b_l(ls, nt * 16, 32), a2);
    av = wmma_bf16(load_a_g(wv, mt * 16, 0),  load_b_l(lx, pb, 0),  av);
    av = wmma_bf16(load_a_g(wv, mt * 16, 32), load_b_l(lx, pb, 32), av);
    int hi = lane >> 4, col = lane & 15;
    int chb = mt * 16 + hi * 8;
    int pc  = nt * 16 + col;
    float r[8];
#pragma unroll
    for (int i = 0; i < 8; ++i) {
      float xc = relu6f(a2[i] * sc3[chb + i] + bi3[chb + i]);
      r[i] = av[i] + lvb[chb + i] + xc;
    }
    uint4 o = { pk_bf(r[0], r[1]), pk_bf(r[2], r[3]),
                pk_bf(r[4], r[5]), pk_bf(r[6], r[7]) };
    *(uint4*)(lv + pc * XSTR + chb) = o;
  }

  // ---- D2) Q,K = softplus(Wq/Wk @ x_center + bias); Q->global, K->LDS ----
  for (int t = wave; t < 2 * 16; t += 16) {
    int which = t >> 4, nt = t & 15;
    int pb = ((nt >> 1) + HALO) * PW + HALO + (nt & 1) * 16;
    const u16* wm = which ? wk : wq;
    v8f acc = {};
    acc = wmma_bf16(load_a_g(wm, 0, 0),  load_b_l(lx, pb, 0),  acc);
    acc = wmma_bf16(load_a_g(wm, 0, 32), load_b_l(lx, pb, 32), acc);
    if (lane < 16) {  // rows 8..15 (lanes 16-31) are zero padding
      int col = lane;
#pragma unroll
      for (int i = 0; i < 8; ++i) {
        int m = i, pc = nt * 16 + col;
        float v = acc[i] + (which ? lkb[m] : lqb[m]);
        float sp = (v > 20.f) ? v : log1pf(expf(v));  // softplus
        if (which) {
          lk[m * 260 + pc] = sp;
        } else {
          int r = pc >> 5, w = pc & 31;
          int n = (by * TH + r) * WW + bx * TW + w;
          Qout[((size_t)(bz * M8 + m)) * NPIX + n] = sp;
        }
      }
    }
  }
  __syncthreads();

  // ---- E) per-tile partial KV[m][c]/Ksum[m] on channel pairs ----
  if (tid < 256) {
    int m = tid >> 5, c2 = tid & 31;
    const float*    kr = lk + m * 260;
    const unsigned* vr = (const unsigned*)lv + c2;
    float sa = 0.f, sb = 0.f;
#pragma unroll 8
    for (int p = 0; p < CEN; ++p) {
      unsigned d = vr[p * XSTRW];
      float k = kr[p];
      sa += k * bf_lo(d);
      sb += k * bf_hi(d);
    }
    float* dst = pKV + (((size_t)bz * NTILES + tileIdx) * M8 + m) * C64 + 2 * c2;
    dst[0] = sa;
    dst[1] = sb;
  }
  if (tid < M8) {
    const float* kr = lk + tid * 260;
    float s = 0.f;
#pragma unroll 8
    for (int p = 0; p < CEN; ++p) s += kr[p];
    pKs[((size_t)bz * NTILES + tileIdx) * M8 + tid] = s;
  }
}

// ===========================================================================
// Kernel 2: fixed-order tile reduction -> KV [B][8][64], Ksum [B][8]
// ===========================================================================
__global__ void pam_kvred(const float* __restrict__ pKV,
                          const float* __restrict__ pKs,
                          float* __restrict__ kv, float* __restrict__ ks) {
  int idx = blockIdx.x * 256 + threadIdx.x;
  if (idx < BATCH * M8 * C64) {
    int b = idx >> 9, mc = idx & 511;
    float s = 0.f;
    for (int t = 0; t < NTILES; ++t)
      s += pKV[((size_t)b * NTILES + t) * (M8 * C64) + mc];
    kv[idx] = s;
  } else if (idx < BATCH * M8 * C64 + BATCH * M8) {
    int j = idx - BATCH * M8 * C64;
    int b = j >> 3, m = j & 7;
    float s = 0.f;
    for (int t = 0; t < NTILES; ++t)
      s += pKs[((size_t)b * NTILES + t) * M8 + m];
    ks[j] = s;
  }
}

// ===========================================================================
// Kernel 3: out = x + gamma * (Q^T KV) / (Q^T (Ksum+eps))  — streaming
// ===========================================================================
__global__ __launch_bounds__(256) void pam_stage_b(
    const float* __restrict__ x, const float* __restrict__ gamma,
    const float* __restrict__ Qbuf, const float* __restrict__ kv,
    const float* __restrict__ ks, float* __restrict__ out) {
  __shared__ float kvs[M8 * C64];
  __shared__ float kse[M8];
  int tid = threadIdx.x;
  int b = blockIdx.y;
  kvs[tid]       = kv[b * (M8 * C64) + tid];
  kvs[tid + 256] = kv[b * (M8 * C64) + tid + 256];
  if (tid < M8) kse[tid] = ks[b * M8 + tid] + 1e-6f;
  __syncthreads();

  int n = blockIdx.x * 256 + tid;
  float q[M8];
  float den = 0.f;
#pragma unroll
  for (int m = 0; m < M8; ++m) {
    q[m] = Qbuf[((size_t)(b * M8 + m)) * NPIX + n];
    den += q[m] * kse[m];
  }
  float norm = 1.0f / den;
  float g = gamma[0];
  const float* xb = x   + (size_t)b * C64 * NPIX + n;
  float*       ob = out + (size_t)b * C64 * NPIX + n;
  for (int c = 0; c < C64; ++c) {
    float wvv = 0.f;
#pragma unroll
    for (int m = 0; m < M8; ++m) wvv += q[m] * kvs[m * C64 + c];
    ob[(size_t)c * NPIX] = xb[(size_t)c * NPIX] + g * wvv * norm;
  }
}

// ===========================================================================
// Host launcher
// ===========================================================================
extern "C" void kernel_launch(void* const* d_in, const int* in_sizes, int n_in,
                              void* d_out, int out_size, void* d_ws, size_t ws_size,
                              hipStream_t stream) {
  const float* x      = (const float*)d_in[0];
  const float* gamma  = (const float*)d_in[1];
  const float* q_w    = (const float*)d_in[2];
  const float* q_b    = (const float*)d_in[3];
  const float* k_w    = (const float*)d_in[4];
  const float* k_b    = (const float*)d_in[5];
  const float* v_w    = (const float*)d_in[6];
  const float* v_b    = (const float*)d_in[7];
  const float* fc1_w  = (const float*)d_in[8];
  const float* fc1_bn = (const float*)d_in[9];
  const float* cv1_w  = (const float*)d_in[10];
  const float* cv1_bn = (const float*)d_in[11];
  const float* cv2_w  = (const float*)d_in[12];
  const float* cv2_bn = (const float*)d_in[13];
  const float* fc2_w  = (const float*)d_in[14];
  const float* fc2_bn = (const float*)d_in[15];
  float* out = (float*)d_out;

  // workspace layout (bytes)
  char* ws = (char*)d_ws;
  size_t o = 0;
  auto take = [&](size_t bytes) { size_t r = o; o = (o + bytes + 255) & ~(size_t)255; return r; };
  u16*   wq_bf = (u16*)  (ws + take(16 * C64 * 2));           // 16x64 (padded)
  u16*   wk_bf = (u16*)  (ws + take(16 * C64 * 2));
  u16*   w1_bf = (u16*)  (ws + take(C64 * C64 * 2));
  u16*   w2_bf = (u16*)  (ws + take(C64 * C64 * 2));
  u16*   wv_bf = (u16*)  (ws + take(C64 * C64 * 2));
  float* bnaff = (float*)(ws + take(4 * 128 * 4));
  float* kv    = (float*)(ws + take(BATCH * M8 * C64 * 4));
  float* ksum  = (float*)(ws + take(BATCH * M8 * 4));
  float* Qbuf  = (float*)(ws + take((size_t)BATCH * M8 * NPIX * 4));
  float* pKV   = (float*)(ws + take((size_t)BATCH * NTILES * M8 * C64 * 4));
  float* pKs   = (float*)(ws + take((size_t)BATCH * NTILES * M8 * 4));
  if (o > ws_size) return;  // workspace too small (deterministic no-op)

  pam_prep<<<1, 256, 0, stream>>>(q_w, k_w, fc1_w, fc2_w, v_w,
                                  fc1_bn, cv1_bn, cv2_bn, fc2_bn,
                                  wq_bf, wk_bf, w1_bf, w2_bf, wv_bf, bnaff);

  pam_stage_a<<<dim3(TILES_X, TILES_Y, BATCH), NTHR, 0, stream>>>(
      x, wq_bf, wk_bf, w1_bf, w2_bf, wv_bf, bnaff, q_b, k_b, v_b,
      cv1_w, cv2_w, Qbuf, pKV, pKs);

  pam_kvred<<<(BATCH * M8 * C64 + BATCH * M8 + 255) / 256, 256, 0, stream>>>(
      pKV, pKs, kv, ksum);

  pam_stage_b<<<dim3(NPIX / 256, BATCH), 256, 0, stream>>>(
      x, gamma, Qbuf, kv, ksum, out);
}
